// DAPTriAttStart_36713380446450
// MI455X (gfx1250) — matrix operand order repaired and checked by reading, not compiled, not run on hardware
//
#include <hip/hip_runtime.h>
#include <hip/hip_bf16.h>

// ---------------------------------------------------------------------------
// Triangle attention (starting node), B=1, N=384, D=128, H=4, CH=32.
// bf16 WMMA (v_wmma_f32_16x16x32_bf16) for all contractions, f32 accumulate.
// ---------------------------------------------------------------------------

typedef __attribute__((ext_vector_type(16))) __bf16 v16bf;
typedef __attribute__((ext_vector_type(8)))  __bf16 v8bf;
typedef __attribute__((ext_vector_type(8)))  float  v8f;
typedef __attribute__((ext_vector_type(4)))  float  v4f;

#define NSEQ   384
#define DMODEL 128
#define NHEAD  4
#define CHEAD  32
#define MPOS   (NSEQ * NSEQ)          // 147456 pair positions
#define NCOLW  528                    // 4*128 (q,k,v,g) + 16 (tri bias, 4 used)
#define SCALE_ATT 0.17677669529663687f  // 1/sqrt(32)
#define INF_  1.0e9f

// ---------------- WMMA operand loaders (wave32, CDNA5 layouts) -------------

// A-matrix 16x32 bf16 from row-major source (row stride ld, in elements).
// lanes 0-15: row r=lane,   K = k0..k0+15
// lanes 16-31: row r=lane-16, K = k0+16..k0+31
static __device__ __forceinline__
v16bf ld_a_rowmajor(const __bf16* M, int row0, int k0, int ld, int lane) {
  const int half = lane >> 4;
  const int r    = lane & 15;
  const __bf16* p = M + (size_t)(row0 + r) * ld + k0;
  v8bf lo = *(const v8bf*)(p + half * 16);
  v8bf hi = *(const v8bf*)(p + half * 16 + 8);
  v16bf a;
#pragma unroll
  for (int t = 0; t < 8; ++t) { a[t] = lo[t]; a[t + 8] = hi[t]; }
  return a;
}

// B-matrix 32x16 bf16 where each output column's K-run is contiguous in
// memory (i.e. source is [col][K], stride ld). lanes 0-15: K k0..k0+15,
// lanes 16-31: K k0+16..k0+31; column = col0 + (lane&15).
static __device__ __forceinline__
v16bf ld_b_colcontig(const __bf16* Mt, int col0, int k0, int ld, int lane) {
  const int c    = col0 + (lane & 15);
  const int koff = k0 + ((lane >> 4) << 4);
  const __bf16* p = Mt + (size_t)c * ld + koff;
  v8bf lo = *(const v8bf*)(p);
  v8bf hi = *(const v8bf*)(p + 8);
  v16bf b;
#pragma unroll
  for (int t = 0; t < 8; ++t) { b[t] = lo[t]; b[t + 8] = hi[t]; }
  return b;
}

static __device__ __forceinline__ v8f wmma_bf16(v16bf a, v16bf b, v8f c) {
  return __builtin_amdgcn_wmma_f32_16x16x32_bf16(
      /*neg_a=*/false, a, /*neg_b=*/false, b,
      /*c_mod=*/(short)0, c, /*reuse_a=*/false, /*reuse_b=*/false);
}

// width-16 reductions (stay inside each lane half of the C layout)
static __device__ __forceinline__ float redmax16(float v) {
#pragma unroll
  for (int m = 1; m <= 8; m <<= 1) v = fmaxf(v, __shfl_xor(v, m, 32));
  return v;
}
static __device__ __forceinline__ float redsum16(float v) {
#pragma unroll
  for (int m = 1; m <= 8; m <<= 1) v += __shfl_xor(v, m, 32);
  return v;
}
static __device__ __forceinline__ float redsum32(float v) {
#pragma unroll
  for (int m = 1; m <= 16; m <<= 1) v += __shfl_xor(v, m, 32);
  return v;
}

// ---------------- Kernel 1: pack/transpose weights to bf16 -----------------
// wt[528][128]: rows 0-127 wq^T | 128-255 wk^T | 256-383 wv^T | 384-511 wg^T
//               512-515 w_bias^T | 516-527 zero
// wot[128][128] = wo^T
__global__ __launch_bounds__(256) void prep_w(
    const float* __restrict__ wq, const float* __restrict__ wk,
    const float* __restrict__ wv, const float* __restrict__ wg,
    const float* __restrict__ wb, const float* __restrict__ wo,
    __bf16* __restrict__ wt, __bf16* __restrict__ wot) {
  int tid = blockIdx.x * blockDim.x + threadIdx.x;
  if (tid < NCOLW * DMODEL) {
    int r = tid / DMODEL, e = tid % DMODEL;
    float v;
    if      (r < 128) v = wq[e * 128 + r];
    else if (r < 256) v = wk[e * 128 + (r - 128)];
    else if (r < 384) v = wv[e * 128 + (r - 256)];
    else if (r < 512) v = wg[e * 128 + (r - 384)];
    else if (r < 516) v = wb[e * NHEAD + (r - 512)];
    else              v = 0.0f;
    wt[tid] = (__bf16)v;
  } else if (tid < NCOLW * DMODEL + DMODEL * DMODEL) {
    int t = tid - NCOLW * DMODEL;
    int d = t / DMODEL, e = t % DMODEL;
    wot[t] = (__bf16)wo[e * DMODEL + d];
  }
}

// ---------------- Kernel 2: LayerNorm -> xn (bf16) -------------------------
// One wave per pair position; each lane owns 4 channels.
__global__ __launch_bounds__(256) void ln_kernel(
    const float* __restrict__ x, const float* __restrict__ lng,
    const float* __restrict__ lnb, __bf16* __restrict__ xn) {
  int lane = threadIdx.x & 31;
  int w    = threadIdx.x >> 5;
  size_t p = (size_t)blockIdx.x * 8 + w;
  v4f xv = *(const v4f*)(x + p * DMODEL + lane * 4);
  float s = xv[0] + xv[1] + xv[2] + xv[3];
  s = redsum32(s);
  float mu = s * (1.0f / DMODEL);
  float d0 = xv[0] - mu, d1 = xv[1] - mu, d2 = xv[2] - mu, d3 = xv[3] - mu;
  float sq = d0 * d0 + d1 * d1 + d2 * d2 + d3 * d3;
  sq = redsum32(sq);
  float rstd = rsqrtf(sq * (1.0f / DMODEL) + 1e-5f);
  float dd[4] = {d0, d1, d2, d3};
#pragma unroll
  for (int t = 0; t < 4; ++t) {
    int c = lane * 4 + t;
    xn[p * DMODEL + c] = (__bf16)(dd[t] * rstd * lng[c] + lnb[c]);
  }
}

// ---------------- Kernel 3: fused QKVG + tri-bias GEMM ---------------------
// xn[147456 x 128] @ wt^T  -> qkvg bf16 [147456 x 512] (+ sigmoid on g cols)
//                          -> tri f32   [4 x 147456]
__global__ __launch_bounds__(256) void proj_gemm(
    const __bf16* __restrict__ xn, const __bf16* __restrict__ wt,
    const float* __restrict__ bg, __bf16* __restrict__ qkvg,
    float* __restrict__ tri) {
  int lane = threadIdx.x & 31;
  int w    = threadIdx.x >> 5;
  int mtile = blockIdx.x * 8 + w;
  int m0 = mtile * 16;
  int half = lane >> 4, lc = lane & 15;

  v16bf a[4];
#pragma unroll
  for (int ks = 0; ks < 4; ++ks) a[ks] = ld_a_rowmajor(xn, m0, ks * 32, DMODEL, lane);

  for (int nt = 0; nt < 33; ++nt) {
    v8f acc = {0.f, 0.f, 0.f, 0.f, 0.f, 0.f, 0.f, 0.f};
#pragma unroll
    for (int ks = 0; ks < 4; ++ks) {
      v16bf b = ld_b_colcontig(wt, nt * 16, ks * 32, DMODEL, lane);
      acc = wmma_bf16(a[ks], b, acc);
    }
    int col = nt * 16 + lc;
#pragma unroll
    for (int j = 0; j < 8; ++j) {
      size_t p = (size_t)m0 + j + 8 * half;
      float v = acc[j];
      if (col < 384) {
        qkvg[p * 512 + col] = (__bf16)v;
      } else if (col < 512) {
        float t = v + bg[col - 384];
        qkvg[p * 512 + col] = (__bf16)(1.0f / (1.0f + __expf(-t)));
      } else if (col < 516) {
        tri[(size_t)(col - 512) * MPOS + p] = v;
      }
    }
  }
}

// ---------------- Kernel 4: fused flash attention + gating -----------------
// grid (qtile=3, head=4, row=384); 8 waves; each wave owns 16 queries.
__global__ __launch_bounds__(256) void attn_kernel(
    const __bf16* __restrict__ qkvg, const float* __restrict__ tri,
    const float* __restrict__ mask, __bf16* __restrict__ og) {
  __shared__ __bf16 lds_p[8][16][32];   // per-wave P-tile staging, 8 KB

  int lane = threadIdx.x & 31;
  int w    = threadIdx.x >> 5;
  int half = lane >> 4, lc = lane & 15;
  int h = blockIdx.y;
  int i = blockIdx.z;
  int q0 = blockIdx.x * 128 + w * 16;           // query-tile base (column n)
  size_t rowbase = (size_t)i * NSEQ;            // pair-row base position

  // Q tile (A operand): rows p = rowbase+q0+r, K = channel c
  v16bf aq;
  {
    size_t p = rowbase + q0 + lc;
    const __bf16* ptr = qkvg + p * 512 + h * CHEAD;
    v8bf lo = *(const v8bf*)(ptr + half * 16);
    v8bf hi = *(const v8bf*)(ptr + half * 16 + 8);
#pragma unroll
    for (int t = 0; t < 8; ++t) { aq[t] = lo[t]; aq[t + 8] = hi[t]; }
  }

  float mrun[8], lrun[8];
#pragma unroll
  for (int j = 0; j < 8; ++j) { mrun[j] = -1e30f; lrun[j] = 0.0f; }
  v8f o0 = {0.f, 0.f, 0.f, 0.f, 0.f, 0.f, 0.f, 0.f};
  v8f o1 = {0.f, 0.f, 0.f, 0.f, 0.f, 0.f, 0.f, 0.f};

  const float* trh = tri + (size_t)h * MPOS;

  for (int mm = 0; mm < NSEQ; mm += 32) {
    // --- scores: Q [16x32] x K^T [32x16] x2 --------------------------------
    v16bf bk0, bk1;
    {
      int m = mm + lc;
      const __bf16* kp0 = qkvg + (rowbase + m) * 512 + 128 + h * CHEAD + half * 16;
      const __bf16* kp1 = qkvg + (rowbase + m + 16) * 512 + 128 + h * CHEAD + half * 16;
      v8bf l0 = *(const v8bf*)(kp0), h0 = *(const v8bf*)(kp0 + 8);
      v8bf l1 = *(const v8bf*)(kp1), h1 = *(const v8bf*)(kp1 + 8);
#pragma unroll
      for (int t = 0; t < 8; ++t) {
        bk0[t] = l0[t]; bk0[t + 8] = h0[t];
        bk1[t] = l1[t]; bk1[t + 8] = h1[t];
      }
    }
    v8f s0 = {0.f, 0.f, 0.f, 0.f, 0.f, 0.f, 0.f, 0.f};
    v8f s1 = {0.f, 0.f, 0.f, 0.f, 0.f, 0.f, 0.f, 0.f};
    s0 = wmma_bf16(aq, bk0, s0);
    s1 = wmma_bf16(aq, bk1, s1);

    float mb0 = INF_ * (mask[mm + lc] - 1.0f);
    float mb1 = INF_ * (mask[mm + 16 + lc] - 1.0f);

    // --- online softmax (rows live on (vgpr j, lane half)) -----------------
#pragma unroll
    for (int j = 0; j < 8; ++j) {
      int q = q0 + j + 8 * half;
      float t0 = s0[j] * SCALE_ATT + trh[(size_t)q * NSEQ + mm + lc] + mb0;
      float t1 = s1[j] * SCALE_ATT + trh[(size_t)q * NSEQ + mm + 16 + lc] + mb1;
      float rm = redmax16(fmaxf(t0, t1));
      float mnew  = fmaxf(mrun[j], rm);
      float alpha = __expf(mrun[j] - mnew);
      float p0 = __expf(t0 - mnew);
      float p1 = __expf(t1 - mnew);
      float rs = redsum16(p0 + p1);
      lrun[j] = lrun[j] * alpha + rs;
      mrun[j] = mnew;
      o0[j] *= alpha;
      o1[j] *= alpha;
      int r = j + 8 * half;
      lds_p[w][r][lc]      = (__bf16)p0;
      lds_p[w][r][16 + lc] = (__bf16)p1;
    }
    // wave-private LDS RAW: drain DS counter (CDNA5 split counters)
    asm volatile("s_wait_dscnt 0x0" ::: "memory");

    // --- P tile back as A operand ------------------------------------------
    v16bf ap;
    {
      const __bf16* pp = &lds_p[w][lc][0];
      v8bf lo = *(const v8bf*)(pp + half * 16);
      v8bf hi = *(const v8bf*)(pp + half * 16 + 8);
#pragma unroll
      for (int t = 0; t < 8; ++t) { ap[t] = lo[t]; ap[t + 8] = hi[t]; }
    }

    // --- V tiles (B operand, K = key index, strided gathers) ---------------
    v16bf bv0, bv1;
    {
      size_t mb = rowbase + mm + half * 16;
      const __bf16* vp = qkvg + mb * 512 + 256 + h * CHEAD;
#pragma unroll
      for (int t = 0; t < 16; ++t) {
        bv0[t] = vp[(size_t)t * 512 + lc];
        bv1[t] = vp[(size_t)t * 512 + 16 + lc];
      }
    }
    o0 = wmma_bf16(ap, bv0, o0);
    o1 = wmma_bf16(ap, bv1, o1);
  }

  // --- epilogue: normalize, gate, store bf16 --------------------------------
#pragma unroll
  for (int j = 0; j < 8; ++j) {
    size_t p = rowbase + q0 + j + 8 * half;
    float inv = 1.0f / lrun[j];
    float g0 = (float)qkvg[p * 512 + 384 + h * CHEAD + lc];
    float g1 = (float)qkvg[p * 512 + 384 + h * CHEAD + 16 + lc];
    og[p * DMODEL + h * CHEAD + lc]      = (__bf16)(o0[j] * inv * g0);
    og[p * DMODEL + h * CHEAD + 16 + lc] = (__bf16)(o1[j] * inv * g1);
  }
}

// ---------------- Kernel 5: output projection ------------------------------
__global__ __launch_bounds__(256) void out_gemm(
    const __bf16* __restrict__ og, const __bf16* __restrict__ wot,
    const float* __restrict__ bo, float* __restrict__ out) {
  int lane = threadIdx.x & 31;
  int w    = threadIdx.x >> 5;
  int mtile = blockIdx.x * 8 + w;
  int m0 = mtile * 16;
  int half = lane >> 4, lc = lane & 15;

  v16bf a[4];
#pragma unroll
  for (int ks = 0; ks < 4; ++ks) a[ks] = ld_a_rowmajor(og, m0, ks * 32, DMODEL, lane);

  for (int nt = 0; nt < 8; ++nt) {
    v8f acc = {0.f, 0.f, 0.f, 0.f, 0.f, 0.f, 0.f, 0.f};
#pragma unroll
    for (int ks = 0; ks < 4; ++ks) {
      v16bf b = ld_b_colcontig(wot, nt * 16, ks * 32, DMODEL, lane);
      acc = wmma_bf16(a[ks], b, acc);
    }
    int col = nt * 16 + lc;
    float bias = bo[col];
#pragma unroll
    for (int j = 0; j < 8; ++j) {
      size_t p = (size_t)m0 + j + 8 * half;
      out[p * DMODEL + col] = acc[j] + bias;
    }
  }
}

// ---------------------------------------------------------------------------
extern "C" void kernel_launch(void* const* d_in, const int* in_sizes, int n_in,
                              void* d_out, int out_size, void* d_ws, size_t ws_size,
                              hipStream_t stream) {
  (void)in_sizes; (void)n_in; (void)out_size; (void)ws_size;
  const float* x    = (const float*)d_in[0];
  const float* mask = (const float*)d_in[1];
  const float* lng  = (const float*)d_in[2];
  const float* lnb  = (const float*)d_in[3];
  const float* wb   = (const float*)d_in[4];
  const float* wq   = (const float*)d_in[5];
  const float* wk   = (const float*)d_in[6];
  const float* wv   = (const float*)d_in[7];
  const float* wg   = (const float*)d_in[8];
  const float* bg   = (const float*)d_in[9];
  const float* wo   = (const float*)d_in[10];
  const float* bo   = (const float*)d_in[11];

  char* ws = (char*)d_ws;
  __bf16* xn   = (__bf16*)(ws);                       // 147456*128*2 = 37,748,736
  __bf16* wt   = (__bf16*)(ws + 37748736);            // 528*128*2    =    135,168
  __bf16* wot  = (__bf16*)(ws + 37883904);            // 128*128*2    =     32,768
  __bf16* qkvg = (__bf16*)(ws + 37916672);            // 147456*512*2 = 150,994,944
  float*  tri  = (float*) (ws + 188911616);           // 4*147456*4   =   2,359,296
  __bf16* og   = (__bf16*)(ws + 191270912);           // 147456*128*2 = 37,748,736

  prep_w<<<((NCOLW + DMODEL) * DMODEL + 255) / 256, 256, 0, stream>>>(
      wq, wk, wv, wg, wb, wo, wt, wot);
  ln_kernel<<<MPOS / 8, 256, 0, stream>>>(x, lng, lnb, xn);
  proj_gemm<<<MPOS / 16 / 8, 256, 0, stream>>>(xn, wt, bg, qkvg, tri);
  attn_kernel<<<dim3(3, NHEAD, NSEQ), 256, 0, stream>>>(qkvg, tri, mask, og);
  out_gemm<<<MPOS / 16 / 8, 256, 0, stream>>>(og, wot, bo, (float*)d_out);
}
